// GATEncoder_39659728011298
// MI455X (gfx1250) — matrix operand order repaired and checked by reading, not compiled, not run on hardware
//
#include <hip/hip_runtime.h>
#include <hip/hip_bf16.h>

#define HEADS 4
#define HID 64
#define HHD 256           // HEADS*HID
#define HALF 256          // IN_DIM/2
#define IN_DIM 512
#define NEG_SLOPE 0.2f
#define BN_EPS 1e-5f

typedef __bf16 bf16_t;
typedef bf16_t bf16x16 __attribute__((ext_vector_type(16)));
typedef bf16_t bf16x8  __attribute__((ext_vector_type(8)));
typedef float  f32x8   __attribute__((ext_vector_type(8)));

union Frag { bf16x16 v; bf16x8 h[2]; };

__device__ __forceinline__ unsigned short f2bf(float f) {
  unsigned int u = __float_as_uint(f);
  u += 0x7fffu + ((u >> 16) & 1u);          // round-to-nearest-even
  return (unsigned short)(u >> 16);
}

// order-preserving float<->int map for atomicMax on floats (involution)
__device__ __forceinline__ int ordmap(float f) {
  int i = __float_as_int(f);
  return i >= 0 ? i : (i ^ 0x7fffffff);
}
__device__ __forceinline__ float ordunmap(int i) {
  return __int_as_float(i >= 0 ? i : (i ^ 0x7fffffff));
}

// ---- CDNA5 async global->LDS copy (ASYNCcnt path), 16B per lane ----------
__device__ __forceinline__ void async_copy_b128(unsigned int lds_byte_addr,
                                                const void* gaddr) {
  asm volatile("global_load_async_to_lds_b128 %0, %1, off"
               :: "v"(lds_byte_addr),
                  "v"((unsigned long long)gaddr)
               : "memory");
}
__device__ __forceinline__ void wait_async0() {
  asm volatile("s_wait_asynccnt 0x0" ::: "memory");
}

// ---------------------------------------------------------------------------
// LDS-staged WMMA GEMM for Ncols == 256:
//   C[M,256] = A_bf16[M,K] @ B_bf16[256,K]^T (+bias)
// One block = one 16-row M-tile; 4 waves cover the 4 column groups of 64 and
// SHARE the A tile: wave 0 stages the 16x32 A slab (1 KB) into LDS with
// async global->LDS copies (double buffered) while all waves consume the
// previous slab via ds_load fragments and issue 4 WMMAs per K-step.
// Requires M%16==0, K%32==0.
// ---------------------------------------------------------------------------
__global__ void __launch_bounds__(128)
gemm_bf16_wmma_ldsA(const unsigned short* __restrict__ A,
                    const unsigned short* __restrict__ B,
                    const float* __restrict__ bias,
                    float* __restrict__ C,
                    int M, int K)
{
  __shared__ unsigned short shA[2][16 * 32];       // 2 x 1 KB
  const int mt   = blockIdx.x;
  const int wv   = threadIdx.x >> 5;               // column group 0..3
  const int lane = threadIdx.x & 31;
  const int l    = lane & 15;
  const int hi   = lane >> 4;
  const int nb   = wv << 6;

  const unsigned int lds0 = (unsigned int)(unsigned long long)&shA[0][0];

  // prologue: stage K-step 0 into buffer 0 (wave 0; 2 x 512B async copies)
  if (threadIdx.x < 32) {
    const int c0 = lane, c1 = lane + 32;           // 16B chunks: r=c>>2, q=c&3
    async_copy_b128(lds0 + c0 * 16,
                    A + (size_t)(mt * 16 + (c0 >> 2)) * K + (c0 & 3) * 8);
    async_copy_b128(lds0 + c1 * 16,
                    A + (size_t)(mt * 16 + (c1 >> 2)) * K + (c1 & 3) * 8);
    wait_async0();
  }
  __syncthreads();

  f32x8 acc[4] = {};
  for (int k0 = 0; k0 < K; k0 += 32) {
    const int p = (k0 >> 5) & 1;
    const bool more = (k0 + 32) < K;
    if (more && threadIdx.x < 32) {                // stage next slab -> buf p^1
      const int kn = k0 + 32;
      const int c0 = lane, c1 = lane + 32;
      const unsigned int ldsn = lds0 + (unsigned)((p ^ 1) << 10);
      async_copy_b128(ldsn + c0 * 16,
                      A + (size_t)(mt * 16 + (c0 >> 2)) * K + kn + (c0 & 3) * 8);
      async_copy_b128(ldsn + c1 * 16,
                      A + (size_t)(mt * 16 + (c1 >> 2)) * K + kn + (c1 & 3) * 8);
    }
    // A fragment from LDS buffer p (row l, 32 K-values, 64B)
    Frag a;
    const unsigned short* sp = &shA[p][l * 32];
    a.h[0] = *(const bf16x8*)(sp + hi * 16);       // K {0..7} / {16..23}
    a.h[1] = *(const bf16x8*)(sp + hi * 16 + 8);   // K {8..15} / {24..31}
#pragma unroll
    for (int j = 0; j < 4; ++j) {
      Frag b;
      const unsigned short* Bp = B + (size_t)(nb + j * 16 + l) * K + k0 + hi * 16;
      b.h[0] = *(const bf16x8*)(Bp);
      b.h[1] = *(const bf16x8*)(Bp + 8);
      acc[j] = __builtin_amdgcn_wmma_f32_16x16x32_bf16(
          false, a.v, false, b.v, (short)0, acc[j], false, false);
    }
    if (more) {
      if (threadIdx.x < 32) wait_async0();         // own copies landed
      __syncthreads();                             // publish buf p^1; retire buf p
    }
  }

#pragma unroll
  for (int j = 0; j < 4; ++j) {
    const int col = nb + j * 16 + l;
    const float bb = bias ? bias[col] : 0.0f;
    float* Crow = C + (size_t)(mt * 16 + hi * 8) * 256 + col;
#pragma unroll
    for (int r = 0; r < 8; ++r)
      Crow[(size_t)r * 256] = acc[j][r] + bb;      // VGPR r -> M=r (+8 hi lanes)
  }
}

// ---------------------------------------------------------------------------
// Register-direct WMMA GEMM (used for Ncols == 64): one wave per 16x64 strip.
// ---------------------------------------------------------------------------
__global__ void __launch_bounds__(128)
gemm_bf16_wmma(const unsigned short* __restrict__ A,
               const unsigned short* __restrict__ B,
               const float* __restrict__ bias,
               float* __restrict__ C,
               int M, int Ncols, int K)
{
  const int groups = (M >> 4) * (Ncols >> 6);
  const int wid = blockIdx.x * (blockDim.x >> 5) + (threadIdx.x >> 5);
  if (wid >= groups) return;
  const int ng4  = Ncols >> 6;
  const int mt   = wid / ng4;
  const int nb   = (wid % ng4) << 6;
  const int lane = threadIdx.x & 31;
  const int l    = lane & 15;
  const int hi   = lane >> 4;

  const unsigned short* Arow = A + (size_t)(mt * 16 + l) * K;
  f32x8 acc[4] = {};

  for (int k0 = 0; k0 < K; k0 += 32) {
    Frag a;
    a.h[0] = *(const bf16x8*)(Arow + k0 + hi * 16);
    a.h[1] = *(const bf16x8*)(Arow + k0 + hi * 16 + 8);
#pragma unroll
    for (int j = 0; j < 4; ++j) {
      Frag b;
      const unsigned short* Bp = B + (size_t)(nb + j * 16 + l) * K + k0 + hi * 16;
      b.h[0] = *(const bf16x8*)(Bp);
      b.h[1] = *(const bf16x8*)(Bp + 8);
      acc[j] = __builtin_amdgcn_wmma_f32_16x16x32_bf16(
          false, a.v, false, b.v, (short)0, acc[j], false, false);
    }
  }

#pragma unroll
  for (int j = 0; j < 4; ++j) {
    const int col = nb + j * 16 + l;
    const float bb = bias ? bias[col] : 0.0f;
    float* Crow = C + (size_t)(mt * 16 + hi * 8) * Ncols + col;
#pragma unroll
    for (int r = 0; r < 8; ++r)
      Crow[(size_t)r * Ncols] = acc[j][r] + bb;
  }
}

// --------------------------- utility kernels -------------------------------
__global__ void cvt_f32_bf16(const float* __restrict__ in,
                             unsigned short* __restrict__ out, long n) {
  long i = (long)blockIdx.x * blockDim.x + threadIdx.x;
  if (i < n) out[i] = f2bf(in[i]);
}

__global__ void relu_to_bf16(const float* __restrict__ in,
                             unsigned short* __restrict__ out, long n) {
  long i = (long)blockIdx.x * blockDim.x + threadIdx.x;
  if (i < n) out[i] = f2bf(fmaxf(in[i], 0.0f));
}

__global__ void fill_u32(unsigned int* __restrict__ p, unsigned int v, long n) {
  long i = (long)blockIdx.x * blockDim.x + threadIdx.x;
  if (i < n) p[i] = v;
}

// --------------------------- encoder BatchNorm -----------------------------
__global__ void bn_stats(const float* __restrict__ x,
                         float* __restrict__ gsum, float* __restrict__ gsq, int n) {
  const int col = threadIdx.x;               // 256 threads: one column each
  const int r0 = blockIdx.x * 256;
  float s = 0.f, q = 0.f;
  for (int i = 0; i < 256; ++i) {
    int r = r0 + i;
    if (r < n) { float v = x[(size_t)r * HALF + col]; s += v; q += v * v; }
  }
  atomicAdd(&gsum[col], s);
  atomicAdd(&gsq[col], q);
}

__global__ void bn_apply_relu(const float* __restrict__ x,
                              const float* __restrict__ gsum,
                              const float* __restrict__ gsq,
                              const float* __restrict__ gamma,
                              const float* __restrict__ beta,
                              unsigned short* __restrict__ out, int n) {
  long i = (long)blockIdx.x * blockDim.x + threadIdx.x;
  if (i >= (long)n * HALF) return;
  int col = (int)(i & (HALF - 1));
  float inv_n = 1.0f / (float)n;
  float mu = gsum[col] * inv_n;
  float var = gsq[col] * inv_n - mu * mu;
  float v = (x[i] - mu) * rsqrtf(var + BN_EPS) * gamma[col] + beta[col];
  out[i] = f2bf(fmaxf(v, 0.0f));
}

// --------------------------- GAT attention ---------------------------------
__global__ void attn_el_er(const float* __restrict__ ft,
                           const float* __restrict__ al,
                           const float* __restrict__ ar,
                           float* __restrict__ el, float* __restrict__ er, int n) {
  int i = blockIdx.x * blockDim.x + threadIdx.x;
  if (i >= n * HEADS) return;
  int node = i >> 2, h = i & 3;
  const float* f = ft + (size_t)node * HHD + h * HID;
  const float* a = al + h * HID;
  const float* b = ar + h * HID;
  float s0 = 0.f, s1 = 0.f;
#pragma unroll 8
  for (int d = 0; d < HID; ++d) { float v = f[d]; s0 += v * a[d]; s1 += v * b[d]; }
  el[i] = s0; er[i] = s1;
}

__global__ void edge_logits_max(const int* __restrict__ src, const int* __restrict__ dst,
                                const float* __restrict__ el, const float* __restrict__ er,
                                float* __restrict__ wedge, int* __restrict__ emax, int e4) {
  int i = blockIdx.x * blockDim.x + threadIdx.x;
  if (i >= e4) return;
  int e = i >> 2, h = i & 3;
  int s = src[e], d = dst[e];
  float x = el[s * 4 + h] + er[d * 4 + h];
  x = x > 0.f ? x : NEG_SLOPE * x;           // leaky relu
  wedge[i] = x;
  atomicMax(&emax[d * 4 + h], ordmap(x));
}

__global__ void edge_exp_sum(const int* __restrict__ dst,
                             float* __restrict__ wedge, const int* __restrict__ emax,
                             float* __restrict__ denom, int e4) {
  int i = blockIdx.x * blockDim.x + threadIdx.x;
  if (i >= e4) return;
  int e = i >> 2, h = i & 3;
  int d = dst[e];
  float em = ordunmap(emax[d * 4 + h]);
  float w = __expf(wedge[i] - em);
  wedge[i] = w;
  atomicAdd(&denom[d * 4 + h], w);
}

// one thread per (edge, head, 4-wide d-chunk): float4 gather, 4 atomic adds.
// ft and acc both fit in the 192 MB L2 -> atomics stay on-chip.
__global__ void edge_message(const int* __restrict__ src, const int* __restrict__ dst,
                             const float* __restrict__ wedge, const float* __restrict__ denom,
                             const float* __restrict__ ft, float* __restrict__ acc, int E) {
  long i = (long)blockIdx.x * blockDim.x + threadIdx.x;
  if (i >= (long)E * 64) return;
  int e = (int)(i >> 6);
  int t = (int)(i & 63);
  int h = t >> 4;
  int q = (t & 15) << 2;
  int s = src[e], d = dst[e];
  float a = wedge[e * 4 + h] / fmaxf(denom[d * 4 + h], 1e-9f);
  const float4 v = *(const float4*)(ft + (size_t)s * HHD + h * HID + q);
  float* o = acc + (size_t)d * HHD + h * HID + q;
  atomicAdd(o + 0, a * v.x);
  atomicAdd(o + 1, a * v.y);
  atomicAdd(o + 2, a * v.z);
  atomicAdd(o + 3, a * v.w);
}

__global__ void head_mean(const float* __restrict__ acc, float* __restrict__ out, int n) {
  int i = blockIdx.x * blockDim.x + threadIdx.x;
  if (i >= n * HID) return;
  int node = i / HID, d = i - node * HID;
  const float* a = acc + (size_t)node * HHD + d;
  out[i] = 0.25f * (a[0] + a[64] + a[128] + a[192]);
}

// ---------------------------------------------------------------------------
extern "C" void kernel_launch(void* const* d_in, const int* in_sizes, int n_in,
                              void* d_out, int out_size, void* d_ws, size_t ws_size,
                              hipStream_t stream) {
  const float* h      = (const float*)d_in[0];
  const int*   src    = (const int*)d_in[1];
  const int*   dst    = (const int*)d_in[2];
  const float* enc_w1 = (const float*)d_in[3];
  const float* enc_b1 = (const float*)d_in[4];
  const float* gamma  = (const float*)d_in[5];
  const float* beta   = (const float*)d_in[6];
  const float* enc_w2 = (const float*)d_in[7];
  const float* enc_b2 = (const float*)d_in[8];
  const float* W0     = (const float*)d_in[9];
  const float* al0    = (const float*)d_in[10];
  const float* ar0    = (const float*)d_in[11];
  const float* Wm     = (const float*)d_in[12];
  const float* alm    = (const float*)d_in[13];
  const float* arm    = (const float*)d_in[14];

  const int N = in_sizes[0] / IN_DIM;   // 50000 (multiple of 16)
  const int E = in_sizes[1];            // 800000

  // ---- workspace carving (256B aligned, with region reuse) ----
  size_t off = 0;
  auto alloc = [&](size_t b) -> char* {
    char* p = (char*)d_ws + off;
    off += (b + 255) & ~(size_t)255;
    return p;
  };
  unsigned short* w1b = (unsigned short*)alloc((size_t)HALF * IN_DIM * 2);
  unsigned short* w2b = (unsigned short*)alloc((size_t)HID * HALF * 2);
  unsigned short* W0b = (unsigned short*)alloc((size_t)HHD * HID * 2);
  unsigned short* Wmb = (unsigned short*)alloc((size_t)4 * HHD * HHD * 2);

  // region1: bf16 copy of h during the encoder; edge/attention scratch after
  size_t o1 = 0;
  auto carve = [&](size_t b) { size_t p = o1; o1 += (b + 255) & ~(size_t)255; return p; };
  size_t wedge_o = carve((size_t)E * HEADS * 4);
  size_t el_o    = carve((size_t)N * HEADS * 4);
  size_t er_o    = carve((size_t)N * HEADS * 4);
  size_t emax_o  = carve((size_t)N * HEADS * 4);
  size_t den_o   = carve((size_t)N * HEADS * 4);
  size_t gst_o   = carve(512 * 4);
  size_t hbytes  = (size_t)N * IN_DIM * 2;
  char* R1 = alloc(o1 > hbytes ? o1 : hbytes);
  unsigned short* hbf = (unsigned short*)R1;
  float* wedge = (float*)(R1 + wedge_o);
  float* el    = (float*)(R1 + el_o);
  float* er    = (float*)(R1 + er_o);
  int*   emax  = (int*)(R1 + emax_o);
  float* denom = (float*)(R1 + den_o);
  float* gsum  = (float*)(R1 + gst_o);
  float* gsq   = gsum + 256;

  float* x1  = (float*)alloc((size_t)N * HHD * 4);   // encoder hidden, then ft
  float* acc = (float*)alloc((size_t)N * HHD * 4);   // enc2 out, then GAT accumulator
  unsigned short* xbf = (unsigned short*)alloc((size_t)N * HHD * 2); // bf16 activations

  auto blocks = [](long n, long per) { return (unsigned)((n + per - 1) / per); };

  // ---- bf16 conversions (weights + input features) ----
  cvt_f32_bf16<<<blocks((long)HALF * IN_DIM, 256), 256, 0, stream>>>(enc_w1, w1b, (long)HALF * IN_DIM);
  cvt_f32_bf16<<<blocks((long)HID * HALF, 256), 256, 0, stream>>>(enc_w2, w2b, (long)HID * HALF);
  cvt_f32_bf16<<<blocks((long)HHD * HID, 256), 256, 0, stream>>>(W0, W0b, (long)HHD * HID);
  cvt_f32_bf16<<<blocks((long)4 * HHD * HHD, 256), 256, 0, stream>>>(Wm, Wmb, (long)4 * HHD * HHD);
  cvt_f32_bf16<<<blocks((long)N * IN_DIM, 256), 256, 0, stream>>>(h, hbf, (long)N * IN_DIM);

  // ---- encoder: Linear(512->256) -> BN -> ReLU -> Linear(256->64) ----
  gemm_bf16_wmma_ldsA<<<N / 16, 128, 0, stream>>>(hbf, w1b, enc_b1, x1, N, IN_DIM);
  fill_u32<<<2, 256, 0, stream>>>((unsigned int*)gsum, 0u, 512);
  bn_stats<<<blocks(N, 256), 256, 0, stream>>>(x1, gsum, gsq, N);
  bn_apply_relu<<<blocks((long)N * HALF, 256), 256, 0, stream>>>(x1, gsum, gsq, gamma, beta, xbf, N);
  {
    long g = (long)(N / 16) * (HID / 64);
    gemm_bf16_wmma<<<blocks(g, 4), 128, 0, stream>>>(xbf, w2b, enc_b2, acc, N, HID, HALF);
  }
  cvt_f32_bf16<<<blocks((long)N * HID, 256), 256, 0, stream>>>(acc, xbf, (long)N * HID);

  // ---- 5 GAT layers (last one: no relu, mean over heads) ----
  for (int L = 0; L < 5; ++L) {
    const unsigned short* Wb = (L == 0) ? W0b : (Wmb + (size_t)(L - 1) * HHD * HHD);
    const int Kin = (L == 0) ? HID : HHD;
    const float* al = (L == 0) ? al0 : (alm + (size_t)(L - 1) * HEADS * HID);
    const float* ar = (L == 0) ? ar0 : (arm + (size_t)(L - 1) * HEADS * HID);

    gemm_bf16_wmma_ldsA<<<N / 16, 128, 0, stream>>>(xbf, Wb, nullptr, x1, N, Kin);

    attn_el_er<<<blocks((long)N * HEADS, 256), 256, 0, stream>>>(x1, al, ar, el, er, N);
    fill_u32<<<blocks((long)N * HEADS, 256), 256, 0, stream>>>((unsigned int*)emax, 0x807fffffu, (long)N * HEADS); // ordmap(-inf)
    fill_u32<<<blocks((long)N * HEADS, 256), 256, 0, stream>>>((unsigned int*)denom, 0u, (long)N * HEADS);
    fill_u32<<<blocks((long)N * HHD, 256), 256, 0, stream>>>((unsigned int*)acc, 0u, (long)N * HHD);

    edge_logits_max<<<blocks((long)E * HEADS, 256), 256, 0, stream>>>(src, dst, el, er, wedge, emax, E * HEADS);
    edge_exp_sum<<<blocks((long)E * HEADS, 256), 256, 0, stream>>>(dst, wedge, emax, denom, E * HEADS);
    edge_message<<<blocks((long)E * 64, 256), 256, 0, stream>>>(src, dst, wedge, denom, x1, acc, E);

    if (L < 4)
      relu_to_bf16<<<blocks((long)N * HHD, 256), 256, 0, stream>>>(acc, xbf, (long)N * HHD);
    else
      head_mean<<<blocks((long)N * HID, 256), 256, 0, stream>>>(acc, (float*)d_out, N);
  }
  (void)n_in; (void)out_size; (void)ws_size;
}
